// WaveRNN_10075993276494
// MI455X (gfx1250) — matrix operand order repaired and checked by reading, not compiled, not run on hardware
//
// WaveRNN sampling on MI455X (gfx1250): persistent multi-WG kernel, bf16 WMMA.
//
// The T=2048 scan is strictly sequential (argmax feedback) -> latency-bound.
// Per step: 77M-MAC hidden GEMM split over 12 WGPs (bf16 weights, 4.8MB,
// L2-resident), then a small serial GRU+MLP+argmax section inside WG0's WGP
// (LDS staged) so only 2 device-wide barriers are needed per step.
// hiddenT staged to LDS via global_load_async_to_lds_b128 (ASYNCcnt path).
// GEMM k-loops use chunked, sched_barrier-pinned A double-buffering so WMMAs
// overlap outstanding global loads; each wave computes both N-tiles of its
// M-tile (2 accumulators) for 2x A reuse.
#include <hip/hip_runtime.h>
#include <stdint.h>

#define HALF   448
#define BINS   256
#define BATCH  32
#define TSTEPS 2048
#define K2H    896              // 2*HALF
#define MHP    2688             // 6*HALF
#define TB     (TSTEPS*BATCH)
#define INVBIN (2.0f/255.0f)
#define NWG    12
#define NTH    512
#define NWAVE  16
#define KT_HP  28               // 896/32
#define MT_PER_WG 14            // (2688/16)/12
#define HSTRIDE 904             // padded hiddenT LDS stride (452 dw % 64 = 4)
#define POOL_BYTES 62464        // max(32*904*2, 32*260*4 + 32*456*2)

typedef __bf16 bf16;
typedef bf16  v16bf __attribute__((ext_vector_type(16)));
typedef bf16  v8bf  __attribute__((ext_vector_type(8)));
typedef float v8f   __attribute__((ext_vector_type(8)));
typedef float v4f   __attribute__((ext_vector_type(4)));
typedef int   v4i   __attribute__((ext_vector_type(4)));

#if __has_builtin(__builtin_amdgcn_global_load_async_to_lds_b128)
#define HAVE_ASYNC_LDS 1
typedef __attribute__((address_space(1))) v4i* gv4i_p;
typedef __attribute__((address_space(3))) v4i* lv4i_p;
#endif

// ---------------------------------------------------------------- helpers ---
__device__ __forceinline__ v8f wmma_bf16(v16bf a, v16bf b, v8f c) {
  // D = A(16x32 bf16) * B(32x16 bf16) + C(f32)
  return __builtin_amdgcn_wmma_f32_16x16x32_bf16(false, a, false, b,
                                                 (short)0, c, false, false);
}

__device__ __forceinline__ void sched_fence() {
#if __has_builtin(__builtin_amdgcn_sched_barrier)
  __builtin_amdgcn_sched_barrier(0);
#endif
}

// Load one lane's 16 bf16 (two 16B chunks; 16B alignment guaranteed by layout)
__device__ __forceinline__ v16bf load_ab(const bf16* p) {
  v8bf lo = *(const v8bf*)p;
  v8bf hi = *(const v8bf*)(p + 8);
  v16bf r;
#pragma unroll
  for (int i = 0; i < 8; ++i) { r[i] = lo[i]; r[i + 8] = hi[i]; }
  return r;
}

// One M-tile x both N-tiles, K = CH*NC k-tiles. A (global, packed) is
// double-buffered in register chunks of CH tiles; sched_barrier(0) pins the
// "issue chunk c+1 loads" group ahead of the "WMMA chunk c" group so a whole
// chunk of global loads is always in flight behind the WMMAs. B from LDS.
template <int CH, int NC>
__device__ __forceinline__ void gemm2(const bf16* ap, const bf16* b0p,
                                      const bf16* b1p, v8f& acc0, v8f& acc1) {
  v16bf abuf[2][CH];
#pragma unroll
  for (int j = 0; j < CH; ++j) abuf[0][j] = load_ab(ap + j * 512);
  ap += CH * 512;
#pragma unroll
  for (int c = 0; c < NC; ++c) {
    const int cur = c & 1, nxt = cur ^ 1;
    if (c + 1 < NC) {
#pragma unroll
      for (int j = 0; j < CH; ++j) abuf[nxt][j] = load_ab(ap + j * 512);
      ap += CH * 512;
    }
    sched_fence();                       // loads(c+1) stay above wmmas(c)
#pragma unroll
    for (int j = 0; j < CH; ++j) {
      v16bf b0 = load_ab(b0p); b0p += 32;
      v16bf b1 = load_ab(b1p); b1p += 32;
      acc0 = wmma_bf16(abuf[cur][j], b0, acc0);
      acc1 = wmma_bf16(abuf[cur][j], b1, acc1);
    }
    sched_fence();
  }
}

__device__ __forceinline__ float sigmf(float x) { return 1.0f / (1.0f + __expf(-x)); }
__device__ __forceinline__ float tanhfast(float x) {
  float e = __expf(-2.0f * fabsf(x));
  float r = (1.0f - e) / (1.0f + e);
  return copysignf(r, x);
}

// Device-wide sense barrier (state in L2 scratch; reset by prologue each call)
__device__ __forceinline__ void grid_sync(int* bar, int target) {
  __syncthreads();
  if (threadIdx.x == 0) {
    __threadfence();
    int prev = __hip_atomic_fetch_add(&bar[0], 1, __ATOMIC_ACQ_REL,
                                      __HIP_MEMORY_SCOPE_AGENT);
    if (prev == NWG - 1) {
      __hip_atomic_store(&bar[0], 0, __ATOMIC_RELAXED, __HIP_MEMORY_SCOPE_AGENT);
      __hip_atomic_fetch_add(&bar[1], 1, __ATOMIC_ACQ_REL, __HIP_MEMORY_SCOPE_AGENT);
    } else {
      while (__hip_atomic_load(&bar[1], __ATOMIC_ACQUIRE,
                               __HIP_MEMORY_SCOPE_AGENT) < target) {
        __builtin_amdgcn_s_sleep(1);
      }
    }
    __threadfence();
  }
  __syncthreads();
}

// ------------------------------------------------------------- prologue -----
// Pack row-major f32 [M][K] into bf16 A tiles in this file's contiguous k order (k = 16h + s, as B reads):
// tile(mt,kt) = 512 contiguous elems; lane L<16 holds row mt*16+L,
// K = {0..15}; lane L>=16 holds K = {16..31}.
__device__ __forceinline__ void pack_one(const float* src, bf16* dst,
                                         int M, int K, long gid, long stride) {
  const int KT = K >> 5;
  const long total = (long)M * K;
  for (long p = gid; p < total; p += stride) {
    int e    = (int)(p & 15);
    int lane = (int)((p >> 4) & 31);
    long tile = p >> 9;
    int kt = (int)(tile % KT), mt = (int)(tile / KT);
    int m = mt * 16 + (lane & 15);
    int k = (lane < 16) ? e
                        : (16 + e);
    k += kt * 32;
    dst[p] = (bf16)src[(long)m * K + k];
  }
}

__global__ void wavernn_prologue(const float* wh, const float* w1c, const float* w2c,
                                 const float* w1f, const float* w2f,
                                 bf16* whp, bf16* w1cp, bf16* w2cp,
                                 bf16* w1fp, bf16* w2fp,
                                 bf16* hidT, float* chs, float* fhs, int* bar) {
  long gid = (long)blockIdx.x * blockDim.x + threadIdx.x;
  long stride = (long)gridDim.x * blockDim.x;
  pack_one(wh,  whp,  MHP,  K2H,  gid, stride);
  pack_one(w1c, w1cp, HALF, HALF, gid, stride);
  pack_one(w2c, w2cp, BINS, HALF, gid, stride);
  pack_one(w1f, w1fp, HALF, HALF, gid, stride);
  pack_one(w2f, w2fp, BINS, HALF, gid, stride);
  for (long i = gid; i < (long)BATCH * K2H; i += stride) hidT[i] = (bf16)0.0f;
  for (long i = gid; i < (long)HALF * BATCH; i += stride) { chs[i] = 0.f; fhs[i] = 0.f; }
  if (gid == 0) { bar[0] = 0; bar[1] = 0; }
}

// -------------------------------------------------- WG0 serial components ---
// relu(W1[448x448] @ hT + b1) -> G1 ; W2[256x448] @ G1 + b2 -> s_logT[32][260]
__device__ void mlp_to_bins(const bf16* w1p, const float* b1,
                            const bf16* w2p, const float* b2,
                            const bf16* s_hidT, bf16* s_g1T, float* s_logT) {
  const int tid = threadIdx.x;
  const int wave = tid >> 5, lane = tid & 31;
  const int l16 = lane & 15;
  const int hi16 = (lane & 16);       // B-tile K offset: 0 / 16
  const int hi8  = (lane & 16) >> 1;  // D-tile row offset: 0 / 8

  for (int m = wave; m < 28; m += NWAVE) {            // stage 1: 28 mt, both nt
    v8f acc0, acc1;
#pragma unroll
    for (int i = 0; i < 8; ++i) { acc0[i] = 0.f; acc1[i] = 0.f; }
    const bf16* ap  = w1p + ((size_t)(m * 14) << 9) + lane * 16;
    const bf16* b0p = s_hidT + l16 * 456 + hi16;
    const bf16* b1p = s_hidT + (16 + l16) * 456 + hi16;
    gemm2<2, 7>(ap, b0p, b1p, acc0, acc1);            // K = 14 k-tiles

    int row0 = m * 16 + hi8;
    const v4f* bb = (const v4f*)(b1 + row0);
    v4f b0 = bb[0], b4 = bb[1];
    v8bf g0, g1;
#pragma unroll
    for (int i = 0; i < 4; ++i) {
      g0[i]     = (bf16)fmaxf(acc0[i]     + b0[i], 0.f);
      g0[i + 4] = (bf16)fmaxf(acc0[i + 4] + b4[i], 0.f);
      g1[i]     = (bf16)fmaxf(acc1[i]     + b0[i], 0.f);
      g1[i + 4] = (bf16)fmaxf(acc1[i + 4] + b4[i], 0.f);
    }
    *(v8bf*)(s_g1T + l16 * 456 + row0) = g0;
    *(v8bf*)(s_g1T + (16 + l16) * 456 + row0) = g1;
  }
  __syncthreads();

  for (int m = wave; m < 16; m += NWAVE) {            // stage 2: 16 mt, both nt
    v8f acc0, acc1;
#pragma unroll
    for (int i = 0; i < 8; ++i) { acc0[i] = 0.f; acc1[i] = 0.f; }
    const bf16* ap  = w2p + ((size_t)(m * 14) << 9) + lane * 16;
    const bf16* b0p = s_g1T + l16 * 456 + hi16;
    const bf16* b1p = s_g1T + (16 + l16) * 456 + hi16;
    gemm2<2, 7>(ap, b0p, b1p, acc0, acc1);

    int row0 = m * 16 + hi8;
    const v4f* bb = (const v4f*)(b2 + row0);
    v4f b0 = bb[0], b4 = bb[1];
    v4f o00, o04, o10, o14;
#pragma unroll
    for (int i = 0; i < 4; ++i) {
      o00[i] = acc0[i] + b0[i]; o04[i] = acc0[i + 4] + b4[i];
      o10[i] = acc1[i] + b0[i]; o14[i] = acc1[i + 4] + b4[i];
    }
    v4f* d0 = (v4f*)(s_logT + l16 * 260 + row0);
    v4f* d1 = (v4f*)(s_logT + (16 + l16) * 260 + row0);
    d0[0] = o00; d0[1] = o04;
    d1[0] = o10; d1[1] = o14;
  }
  __syncthreads();
}

// argmax over 256 bins per batch (first-max tie rule like jnp.argmax)
__device__ void argmax_bins(const float* s_logT, int* s_idx, float* out_row, int t) {
  const int tid = threadIdx.x, wave = tid >> 5, lane = tid & 31;
  for (int b = wave; b < BATCH; b += NWAVE) {
    float bv = -3.4e38f; int bi = 0;
    const float* p = s_logT + b * 260;
#pragma unroll
    for (int j = 0; j < 8; ++j) {
      int idx = lane + (j << 5);
      float v = p[idx];
      if (v > bv) { bv = v; bi = idx; }    // ascending idx: '>' keeps first max
    }
#pragma unroll
    for (int off = 16; off > 0; off >>= 1) {
      float ov = __shfl_xor(bv, off);
      int   oi = __shfl_xor(bi, off);
      if (ov > bv || (ov == bv && oi < bi)) { bv = ov; bi = oi; }
    }
    if (lane == 0) {
      s_idx[b] = bi;
      out_row[(size_t)b * TSTEPS + t] = (float)bi;
    }
  }
}

// ----------------------------------------------------------- main kernel ----
__global__ void __launch_bounds__(NTH, 1)
wavernn_main(const float* bh, const float* cb, const float* fbias,
             const float* ciw, const float* fiw,
             const float* b1c, const float* b2c, const float* b1f, const float* b2f,
             const bf16* whp, const bf16* w1cp, const bf16* w2cp,
             const bf16* w1fp, const bf16* w2fp,
             bf16* hidT_g, float* ch_state, float* fh_state, float* hpT,
             int* bar, float* out) {
  // One time-multiplexed LDS pool (61 KB):
  //   Phase A : s_hidB = padded hiddenT panel, bf16 [32][904]
  //   Phase BC: s_hidT bf16 [32][456] / s_logT f32 [32][260] (same bytes)
  //             s_g1T  bf16 [32][456] at byte offset 33280
  __shared__ __attribute__((aligned(32))) unsigned char s_pool[POOL_BYTES];
  __shared__ float s_cin0[BATCH], s_cin1[BATCH];
  __shared__ int s_ic[BATCH], s_if[BATCH];
  bf16*  s_hidB = (bf16*)s_pool;
  bf16*  s_hidT = (bf16*)s_pool;
  float* s_logT = (float*)s_pool;
  bf16*  s_g1T  = (bf16*)(s_pool + 33280);

  const int tid = threadIdx.x;
  const int wg = blockIdx.x;
  const int wave = tid >> 5, lane = tid & 31;
  const int l16 = lane & 15;
  const int hi16 = (lane & 16);
  const int hi8  = (lane & 16) >> 1;
  int target = 0;

  if (wg == 0 && tid < BATCH) { s_ic[tid] = 128; s_if[tid] = 0; }  // split_signal(0)
  __syncthreads();

  for (int t = 0; t < TSTEPS; ++t) {
    // ---- stage hiddenT[32][896] -> LDS [32][904] (pad kills bank conflicts)
    {
      // 3584 x 16B lane-transfers; 512 threads x 7 iters, no divergence
      for (int j = tid; j < 32 * 112; j += NTH) {
        int b = j / 112, c = j - b * 112;
#ifdef HAVE_ASYNC_LDS
        __builtin_amdgcn_global_load_async_to_lds_b128(
            (gv4i_p)((const char*)hidT_g + b * 1792 + c * 16),
            (lv4i_p)(s_pool + b * 1808 + c * 16), 0, 0);
#else
        *(uint4*)(s_pool + b * 1808 + c * 16) =
            *(const uint4*)((const char*)hidT_g + b * 1792 + c * 16);
#endif
      }
#ifdef HAVE_ASYNC_LDS
#if __has_builtin(__builtin_amdgcn_s_wait_asynccnt)
      __builtin_amdgcn_s_wait_asynccnt(0);
#else
      asm volatile("s_wait_asynccnt 0x0" ::: "memory");
#endif
#endif
      __syncthreads();
    }

    // ---- Phase A (all WGs): hpT[32][2688] = (Wh @ [ch;fh])^T, bias deferred
    // One M-tile per wave (both N-tiles -> 2x A reuse, 56 WMMAs/wave)
    for (int m = wave; m < MT_PER_WG; m += NWAVE) {
      int mt = wg * MT_PER_WG + m;
      v8f acc0, acc1;
#pragma unroll
      for (int i = 0; i < 8; ++i) { acc0[i] = 0.f; acc1[i] = 0.f; }
      const bf16* ap  = whp + ((size_t)(mt * KT_HP) << 9) + lane * 16;
      const bf16* b0p = s_hidB + l16 * HSTRIDE + hi16;
      const bf16* b1p = s_hidB + (16 + l16) * HSTRIDE + hi16;
      gemm2<4, 7>(ap, b0p, b1p, acc0, acc1);          // K = 28 k-tiles

      int row0 = mt * 16 + hi8;
      v4f o00, o04, o10, o14;
#pragma unroll
      for (int i = 0; i < 4; ++i) {
        o00[i] = acc0[i]; o04[i] = acc0[i + 4];
        o10[i] = acc1[i]; o14[i] = acc1[i + 4];
      }
      v4f* d0 = (v4f*)(hpT + (size_t)l16 * MHP + row0);
      v4f* d1 = (v4f*)(hpT + (size_t)(16 + l16) * MHP + row0);
      d0[0] = o00; d0[1] = o04;
      d1[0] = o10; d1[1] = o14;
    }

    grid_sync(bar, ++target);

    // ---- Phase B/C (WG0 only): GRU -> MLP -> argmax, twice (coarse, fine)
    if (wg == 0) {
      if (tid < BATCH) {
        s_cin0[tid] = (float)s_ic[tid] * INVBIN - 1.0f;   // prev coarse sample
        s_cin1[tid] = (float)s_if[tid] * INVBIN - 1.0f;   // prev fine sample
      }
      __syncthreads();

      for (int i = tid; i < HALF * BATCH; i += NTH) {     // coarse GRU
        int b = i & 31, r = i >> 5;
        float hr = hpT[(size_t)b * MHP + r]       + bh[r];
        float hi = hpT[(size_t)b * MHP + r + 448] + bh[r + 448];
        float hn = hpT[(size_t)b * MHP + r + 896] + bh[r + 896];
        float c0 = s_cin0[b], c1 = s_cin1[b];
        float ir  = ciw[2*r]*c0       + ciw[2*r+1]*c1       + cb[(size_t)r*TB       + (size_t)t*BATCH + b];
        float ii  = ciw[2*(r+448)]*c0 + ciw[2*(r+448)+1]*c1 + cb[(size_t)(r+448)*TB + (size_t)t*BATCH + b];
        float inn = ciw[2*(r+896)]*c0 + ciw[2*(r+896)+1]*c1 + cb[(size_t)(r+896)*TB + (size_t)t*BATCH + b];
        float rg = sigmf(ir + hr);
        float ig = sigmf(ii + hi);
        float nn = tanhfast(inn + rg * hn);
        float nv = nn + ig * (ch_state[i] - nn);
        ch_state[i] = nv;
        bf16 bv = (bf16)nv;
        hidT_g[b * K2H + r] = bv;       // for next step's Phase A (all WGs)
        s_hidT[b * 456 + r] = bv;       // for the coarse MLP below
      }
      __syncthreads();

      mlp_to_bins(w1cp, b1c, w2cp, b2c, s_hidT, s_g1T, s_logT);
      argmax_bins(s_logT, s_ic, out, t);
      __syncthreads();

      for (int i = tid; i < HALF * BATCH; i += NTH) {     // fine GRU
        int b = i & 31, r = i >> 5;
        float hr = hpT[(size_t)b * MHP + r + 1344] + bh[r + 1344];
        float hi = hpT[(size_t)b * MHP + r + 1792] + bh[r + 1792];
        float hn = hpT[(size_t)b * MHP + r + 2240] + bh[r + 2240];
        float c0 = s_cin0[b], c1 = s_cin1[b];
        float c2 = (float)s_ic[b] * INVBIN - 1.0f;        // NEW coarse sample
        float ir  = fiw[3*r]*c0       + fiw[3*r+1]*c1       + fiw[3*r+2]*c2       + fbias[(size_t)r*TB       + (size_t)t*BATCH + b];
        float ii  = fiw[3*(r+448)]*c0 + fiw[3*(r+448)+1]*c1 + fiw[3*(r+448)+2]*c2 + fbias[(size_t)(r+448)*TB + (size_t)t*BATCH + b];
        float inn = fiw[3*(r+896)]*c0 + fiw[3*(r+896)+1]*c1 + fiw[3*(r+896)+2]*c2 + fbias[(size_t)(r+896)*TB + (size_t)t*BATCH + b];
        float rg = sigmf(ir + hr);
        float ig = sigmf(ii + hi);
        float nn = tanhfast(inn + rg * hn);
        float nv = nn + ig * (fh_state[i] - nn);
        fh_state[i] = nv;
        bf16 bv = (bf16)nv;
        hidT_g[b * K2H + 448 + r] = bv;
        s_hidT[b * 456 + r] = bv;
      }
      __syncthreads();

      mlp_to_bins(w1fp, b1f, w2fp, b2f, s_hidT, s_g1T, s_logT);
      argmax_bins(s_logT, s_if, out + (size_t)BATCH * TSTEPS, t);
      __syncthreads();
    }

    grid_sync(bar, ++target);
  }

  if (wg == 0) {   // final hidden states: ch [448][32] f32 then fh
    float* tail = out + (size_t)2 * BATCH * TSTEPS;
    for (int i = tid; i < HALF * BATCH; i += NTH) {
      tail[i] = ch_state[i];
      tail[HALF * BATCH + i] = fh_state[i];
    }
  }
}

// ---------------------------------------------------------------- launch ----
extern "C" void kernel_launch(void* const* d_in, const int* in_sizes, int n_in,
                              void* d_out, int out_size, void* d_ws, size_t ws_size,
                              hipStream_t stream) {
  const float* wh  = (const float*)d_in[0];
  const float* bh  = (const float*)d_in[1];
  const float* cb  = (const float*)d_in[2];
  const float* fb  = (const float*)d_in[3];
  const float* ciw = (const float*)d_in[4];
  const float* fiw = (const float*)d_in[5];
  const float* w1c = (const float*)d_in[6];
  const float* b1c = (const float*)d_in[7];
  const float* w2c = (const float*)d_in[8];
  const float* b2c = (const float*)d_in[9];
  const float* w1f = (const float*)d_in[10];
  const float* b1f = (const float*)d_in[11];
  const float* w2f = (const float*)d_in[12];
  const float* b2f = (const float*)d_in[13];
  float* out = (float*)d_out;

  char* ws = (char*)d_ws;
  size_t off = 0;
  auto take = [&](size_t bytes) {
    char* p = ws + off;
    off = (off + bytes + 255) & ~(size_t)255;
    return p;
  };
  bf16*  whp  = (bf16*)take((size_t)MHP * K2H * 2);
  bf16*  w1cp = (bf16*)take((size_t)HALF * HALF * 2);
  bf16*  w2cp = (bf16*)take((size_t)BINS * HALF * 2);
  bf16*  w1fp = (bf16*)take((size_t)HALF * HALF * 2);
  bf16*  w2fp = (bf16*)take((size_t)BINS * HALF * 2);
  bf16*  hidT = (bf16*)take((size_t)BATCH * K2H * 2);
  float* chs  = (float*)take((size_t)HALF * BATCH * 4);
  float* fhs  = (float*)take((size_t)HALF * BATCH * 4);
  float* hpT  = (float*)take((size_t)BATCH * MHP * 4);
  int*   bar  = (int*)take(256);

  hipLaunchKernelGGL(wavernn_prologue, dim3(1024), dim3(256), 0, stream,
                     wh, w1c, w2c, w1f, w2f,
                     whp, w1cp, w2cp, w1fp, w2fp, hidT, chs, fhs, bar);
  hipLaunchKernelGGL(wavernn_main, dim3(NWG), dim3(NTH), 0, stream,
                     bh, cb, fb, ciw, fiw, b1c, b2c, b1f, b2f,
                     whp, w1cp, w2cp, w1fp, w2fp,
                     hidT, chs, fhs, hpT, bar, out);
}
